// MSDeformAttn_38689065403002
// MI455X (gfx1250) — matrix operand order repaired and checked by reading, not compiled, not run on hardware
//
#include <hip/hip_runtime.h>
#include <hip/hip_bf16.h>

// ---------------- problem constants (from reference) ----------------
#define Dm    256
#define NHD   8
#define NLV   4
#define NPT   4
#define DHd   32          // Dm / NHD, == wave32 lanes (perfect mapping)
#define LQn   19947
#define LINn  19947
#define NBn   2
#define Mrows (NBn * LQn) // 39894

__constant__ int c_H[4]   = {100, 50, 25, 13};
__constant__ int c_W[4]   = {150, 75, 38, 19};
__constant__ int c_OFF[4] = {0, 15000, 18750, 19700};

typedef __attribute__((ext_vector_type(16))) __bf16 v16bf;
typedef __attribute__((ext_vector_type(8)))  float  v8f;

__device__ __forceinline__ unsigned short f2bf(float f) {
    unsigned int u = __builtin_bit_cast(unsigned int, f);
    u += 0x7FFFu + ((u >> 16) & 1u);        // round-to-nearest-even
    return (unsigned short)(u >> 16);
}

// CDNA5 async global->LDS copy (ASYNCcnt path), 16B per lane.
// lds_off is a byte offset from the workgroup LDS base (single __shared__
// allocation => base offset 0).
__device__ __forceinline__ void async_copy_b128(unsigned lds_off, const void* gptr) {
    asm volatile("global_load_async_to_lds_b128 %0, %1, off"
                 :: "v"(lds_off), "v"((unsigned long long)(uintptr_t)gptr)
                 : "memory");
}
__device__ __forceinline__ void wait_async0() {
    asm volatile("s_wait_asynccnt 0x0" ::: "memory");
}

// ---------------- weight prep: fp32 -> bf16, TRANSPOSED [n][k] -------------
__global__ void prep_weights(const float* __restrict__ Wval,
                             const float* __restrict__ Woff,
                             const float* __restrict__ Wattn,
                             const float* __restrict__ Wout,
                             const float* __restrict__ boff,
                             const float* __restrict__ battn,
                             unsigned short* __restrict__ WvT,   // [256][256]
                             unsigned short* __restrict__ WoaT,  // [384][256]
                             unsigned short* __restrict__ WoT,   // [256][256]
                             float* __restrict__ boa) {
    int t = blockIdx.x * blockDim.x + threadIdx.x;
    if (t < 384 * 256) {
        int n = t >> 8, k = t & 255;
        float w = (n < 256) ? Woff[k * 256 + n] : Wattn[k * 128 + (n - 256)];
        WoaT[t] = f2bf(w);
        if (n < 256) {
            WvT[t] = f2bf(Wval[k * 256 + n]);
            WoT[t] = f2bf(Wout[k * 256 + n]);
        }
    }
    if (t < 384) boa[t] = (t < 256) ? boff[t] : battn[t - 256];
}

// ---------------- bf16 WMMA GEMM: C[M x ncols] = A[M x 256] * W^T + bias ---
// Block: 256 threads = 8 waves, tile 128(M) x 64(N), K-step 32, double-
// buffered LDS. Wave w computes rows [16w,16w+16) x 64 cols: one A fragment
// feeds 4 v_wmma_f32_16x16x32_bf16 per k-step. B tiles staged with CDNA5
// async global->LDS b128 copies; A tiles via b128 loads + bf16 convert
// (row-clamped: branch-free, padded rows never stored).
#define ABUF_US  (128 * 40)            // 5120 ushorts / buffer
#define BBUF_US  (64 * 40)             // 2560 ushorts / buffer
#define BUF_US   (ABUF_US + BBUF_US)   // 7680 ushorts = 15360 B / buffer

__global__ __launch_bounds__(256)
void gemm_bf16_wmma(const float* __restrict__ A,
                    const unsigned short* __restrict__ WbT, // bf16 [ncols][256]
                    const float* __restrict__ bias,
                    float* __restrict__ C,
                    int M, int ncols) {
    __shared__ unsigned short smem[2 * BUF_US];   // 30720 B total

    const int tid  = threadIdx.x;
    const int lane = tid & 31, wid = tid >> 5;    // wid = wave's M tile
    const int hi = lane >> 4, lo = lane & 15;
    const int m0 = blockIdx.x * 128;
    const int n0 = blockIdx.y * 64;

    // async B-copy mapping: one b128 (8 bf16) per thread per k-step
    const int bn = tid >> 2;      // 0..63  (column within tile)
    const int bj = tid & 3;       // 0..3   (16B chunk within the 32-K row)
    const unsigned short* bsrc = WbT + (size_t)(n0 + bn) * 256 + bj * 8;

    // A staging mapping: one float4 per thread per iter, 4 iters = 128x32
    v8f acc[4] = {v8f{}, v8f{}, v8f{}, v8f{}};

    auto issueB = [&](int buf, int k0) {
        unsigned lds_off = (unsigned)(buf * (BUF_US * 2) + ABUF_US * 2
                                      + bn * 80 + bj * 16);
        async_copy_b128(lds_off, bsrc + k0);
    };
    auto stageA = [&](int buf, int k0) {
        unsigned short* As = smem + buf * BUF_US;
        #pragma unroll
        for (int i = 0; i < 4; ++i) {
            int id = tid + 256 * i;            // 0..1023
            int m  = id >> 3;                  // 0..127
            int k4 = (id & 7) * 4;             // 0,4,...,28
            int grow = min(m0 + m, M - 1);     // clamp: branch-free
            const float4 a = *(const float4*)(A + (size_t)grow * 256 + k0 + k4);
            unsigned int l32 = (unsigned)f2bf(a.x) | ((unsigned)f2bf(a.y) << 16);
            unsigned int h32 = (unsigned)f2bf(a.z) | ((unsigned)f2bf(a.w) << 16);
            *(uint2*)(As + m * 40 + k4) = make_uint2(l32, h32);  // ds b64
        }
    };
    auto computeTile = [&](int buf) {
        const unsigned short* As = smem + buf * BUF_US;
        const unsigned short* Bs = As + ABUF_US;
        // A fragment (contiguous K order, same as B): row = 16*wid+lo;
        // dword j<4 -> K=16*hi+2j ; dword 4+j -> K=16*hi+8+2j
        union { unsigned int u[8]; v16bf v; } fa, fb[4];
        const unsigned short* ap = &As[(wid * 16 + lo) * 40];
        #pragma unroll
        for (int j = 0; j < 4; ++j) {
            fa.u[j]     = *(const unsigned int*)(ap + 16 * hi + 2 * j);
            fa.u[4 + j] = *(const unsigned int*)(ap + 16 * hi + 8 + 2 * j);
        }
        // B fragments (col = lo + 16*ns, K = 16*hi + e)
        #pragma unroll
        for (int ns = 0; ns < 4; ++ns) {
            const unsigned short* bp = &Bs[(ns * 16 + lo) * 40 + 16 * hi];
            #pragma unroll
            for (int j = 0; j < 8; ++j)
                fb[ns].u[j] = *(const unsigned int*)(bp + 2 * j);
        }
        #pragma unroll
        for (int ns = 0; ns < 4; ++ns)
            acc[ns] = __builtin_amdgcn_wmma_f32_16x16x32_bf16(
                false, fa.v, false, fb[ns].v, (short)0, acc[ns], false, false);
    };

    // software pipeline: prefetch tile 0, then overlap stage(k+1) with mm(k)
    issueB(0, 0);
    stageA(0, 0);
    for (int ki = 0; ki < 8; ++ki) {
        int cur = ki & 1;
        wait_async0();          // our async copy into `cur` is complete
        __syncthreads();        // everyone's staging of `cur` visible
        if (ki < 7) {           // overlap next tile with this tile's WMMA
            issueB(cur ^ 1, (ki + 1) * 32);
            stageA(cur ^ 1, (ki + 1) * 32);
        }
        computeTile(cur);
    }

    // epilogue: C/D layout: col = lo, row = r + 8*hi for VGPR r
    #pragma unroll
    for (int ns = 0; ns < 4; ++ns) {
        int col = n0 + ns * 16 + lo;
        float b = bias[col];
        #pragma unroll
        for (int r = 0; r < 8; ++r) {
            int row = m0 + wid * 16 + r + 8 * hi;
            if (row < M) C[(size_t)row * ncols + col] = acc[ns][r] + b;
        }
    }
}

// ---------------- deformable sampling: one wave per (n,q,h), lane = channel
__global__ __launch_bounds__(256)
void ms_deform_sample(const float* __restrict__ proj,   // [M][384] off|logits
                      const float* __restrict__ refp,   // [M][NL][2]
                      const float* __restrict__ value,  // [NB][LIN][NH][DH]
                      float* __restrict__ outc) {       // [M][NH][DH]
    const int tid  = threadIdx.x;
    const int lane = tid & 31, wid = tid >> 5;
    const int wg = blockIdx.x * 8 + wid;         // 39894*8 == NB*LQ*NH exactly
    const int h  = wg & 7;
    const int nq = wg >> 3;
    const int n  = (nq >= LQn) ? 1 : 0;

    // lanes 0-15 (mirrored on 16-31): one (level,point) each
    const int lp = lane & 15;
    const int l = lp >> 2, p = lp & 3;

    const float* prow = proj + (size_t)nq * 384;
    float logit = prow[256 + h * 16 + lp];
    float offx  = prow[((h * NLV + l) * NPT + p) * 2 + 0];
    float offy  = prow[((h * NLV + l) * NPT + p) * 2 + 1];
    float ref0  = refp[((size_t)nq * NLV + l) * 2 + 0];
    float ref1  = refp[((size_t)nq * NLV + l) * 2 + 1];

    int   Hl = c_H[l], Wl = c_W[l];
    float Hf = (float)Hl, Wf = (float)Wl;
    // reference: loc = ref + off / (H, W); x = loc0*W - .5, y = loc1*H - .5
    float x = (ref0 + offx / Hf) * Wf - 0.5f;
    float y = (ref1 + offy / Wf) * Hf - 0.5f;

    // softmax over the 16 (level,point) logits, width-16 xor reductions
    float mx = logit;
    #pragma unroll
    for (int s = 8; s >= 1; s >>= 1) mx = fmaxf(mx, __shfl_xor(mx, s, 16));
    float e = __expf(logit - mx);
    float ssum = e;
    #pragma unroll
    for (int s = 8; s >= 1; s >>= 1) ssum += __shfl_xor(ssum, s, 16);
    float aw = e / ssum;

    float xf = floorf(x), yf = floorf(y);
    float fx = x - xf, fy = y - yf;
    int x0 = (int)xf, y0 = (int)yf;

    int   rowi[4];
    float wgt[4];                 // products of non-negatives: bits==0 <=> 0.0
    #pragma unroll
    for (int c = 0; c < 4; ++c) {
        int dy = c >> 1, dx = c & 1;
        int yi = y0 + dy, xi = x0 + dx;
        float wc = (dx ? fx : (1.0f - fx)) * (dy ? fy : (1.0f - fy));
        bool valid = (yi >= 0) && (yi < Hl) && (xi >= 0) && (xi < Wl);
        int yc = min(max(yi, 0), Hl - 1);
        int xc = min(max(xi, 0), Wl - 1);
        rowi[c] = c_OFF[l] + yc * Wl + xc;
        wgt[c]  = valid ? (aw * wc) : 0.0f;
    }

    // gather phase: readlane -> SGPR-uniform weight/index, scalar skip branch,
    // lane = channel d => fully coalesced 128B line per gather (L2-resident)
    const float* vbase = value + (size_t)n * LINn * Dm + h * DHd + lane;
    float acc = 0.0f;
    #pragma unroll 1
    for (int s = 0; s < 16; ++s) {
        #pragma unroll
        for (int c = 0; c < 4; ++c) {
            int wbits = __builtin_amdgcn_readlane(
                __builtin_bit_cast(int, wgt[c]), s);
            int rq = __builtin_amdgcn_readlane(rowi[c], s);
            if (wbits != 0) {                 // scalar (SGPR) branch
                float wq = __builtin_bit_cast(float, wbits);
                acc += wq * vbase[(size_t)rq * Dm];
            }
        }
    }
    outc[((size_t)nq * NHD + h) * DHd + lane] = acc;
}

// ---------------- launch ----------------
extern "C" void kernel_launch(void* const* d_in, const int* in_sizes, int n_in,
                              void* d_out, int out_size, void* d_ws, size_t ws_size,
                              hipStream_t stream) {
    const float* query  = (const float*)d_in[0];
    const float* refpts = (const float*)d_in[1];
    const float* inflat = (const float*)d_in[2];
    // d_in[3] = input_spatial_shapes (compile-time constants here)
    const float* W_off  = (const float*)d_in[4];
    const float* b_off  = (const float*)d_in[5];
    const float* W_attn = (const float*)d_in[6];
    const float* b_attn = (const float*)d_in[7];
    const float* W_val  = (const float*)d_in[8];
    const float* b_val  = (const float*)d_in[9];
    const float* W_out  = (const float*)d_in[10];
    const float* b_out  = (const float*)d_in[11];

    char* ws = (char*)d_ws;
    unsigned short* WvT_bf  = (unsigned short*)(ws + 0);          // 131072 B
    unsigned short* WoaT_bf = (unsigned short*)(ws + 131072);     // 196608 B
    unsigned short* WoT_bf  = (unsigned short*)(ws + 327680);     // 131072 B
    float*          bias_oa = (float*)(ws + 458752);              //   1536 B
    float*          value   = (float*)(ws + 460800);              // 40.85 MB
    float*          proj    = (float*)(ws + 41312256);            // 61.28 MB
    float*          outc    = (float*)(ws + 102589440);           // 40.85 MB

    const int M = Mrows;                     // 39894
    const int GM = (M + 127) / 128;          // 312

    prep_weights<<<(384 * 256 + 255) / 256, 256, 0, stream>>>(
        W_val, W_off, W_attn, W_out, b_off, b_attn,
        WvT_bf, WoaT_bf, WoT_bf, bias_oa);

    // value = input_flatten @ W_val + b_val
    gemm_bf16_wmma<<<dim3(GM, 4), 256, 0, stream>>>(inflat, WvT_bf, b_val, value, M, 256);
    // proj = query @ [W_off|W_attn] + [b_off|b_attn]
    gemm_bf16_wmma<<<dim3(GM, 6), 256, 0, stream>>>(query, WoaT_bf, bias_oa, proj, M, 384);
    // softmax + location + bilinear gather -> core output
    ms_deform_sample<<<M, 256, 0, stream>>>(proj, refpts, value, outc);
    // out = core @ W_out + b_out
    gemm_bf16_wmma<<<dim3(GM, 4), 256, 0, stream>>>(outc, WoT_bf, b_out, (float*)d_out, M, 256);
}
